// DynamicConv_87797721465179
// MI455X (gfx1250) — matrix operand order repaired and checked by reading, not compiled, not run on hardware
//
#include <hip/hip_runtime.h>
#include <hip/hip_bf16.h>
#include <hip/hip_fp16.h>

typedef _Float16 v16h __attribute__((ext_vector_type(16)));
typedef _Float16 v8h  __attribute__((ext_vector_type(8)));
typedef _Float16 v2h  __attribute__((ext_vector_type(2)));
typedef float    v8f  __attribute__((ext_vector_type(8)));

#define K_EXP  8
#define D_EMB  128
#define HID    128
#define IN_C   256
#define OUT_C  256
#define BS     32
#define HLEN   4096

__device__ __forceinline__ float gelu_exact(float x) {
    return 0.5f * x * (1.0f + erff(x * 0.70710678118654752440f));
}

// ---------------- Kernel 1: routing MLP -> att[BS][K] ----------------
__global__ void routing_kernel(const float* __restrict__ emb,
                               const float* __restrict__ W0, const float* __restrict__ b0,
                               const float* __restrict__ W1, const float* __restrict__ b1,
                               const float* __restrict__ W2, const float* __restrict__ b2,
                               float* __restrict__ att) {
    __shared__ float s_a[HID];
    __shared__ float s_b[HID];
    __shared__ float s_logit[K_EXP];
    const int b = blockIdx.x, t = threadIdx.x;
    s_a[t] = emb[b * D_EMB + t];
    __syncthreads();
    float acc = b0[t];
    for (int d = 0; d < D_EMB; ++d) acc += s_a[d] * W0[d * HID + t];
    s_b[t] = gelu_exact(acc);
    __syncthreads();
    acc = b1[t];
    for (int d = 0; d < HID; ++d) acc += s_b[d] * W1[d * HID + t];
    const float g = gelu_exact(acc);
    __syncthreads();
    s_a[t] = g;
    __syncthreads();
    if (t < K_EXP) {
        float lg = b2[t];
        for (int d = 0; d < HID; ++d) lg += s_a[d] * W2[d * K_EXP + t];
        s_logit[t] = lg;
    }
    __syncthreads();
    if (t == 0) {
        float m = s_logit[0];
        for (int k = 1; k < K_EXP; ++k) m = fmaxf(m, s_logit[k]);
        float e[K_EXP], sum = 0.f;
        for (int k = 0; k < K_EXP; ++k) { e[k] = expf(s_logit[k] - m); sum += e[k]; }
        const float inv = 1.0f / sum;
        for (int k = 0; k < K_EXP; ++k) att[b * K_EXP + k] = e[k] * inv;
    }
}

// ------- Kernel 2: mix experts -> aggw (f16) [BS][OUT][IN], aggb [BS][OUT] -------
__global__ void agg_kernel(const float* __restrict__ att,
                           const float* __restrict__ weight,
                           const float* __restrict__ bias_k,
                           _Float16* __restrict__ aggw,
                           float* __restrict__ aggb) {
    const int idx = blockIdx.x * blockDim.x + threadIdx.x;   // 0 .. BS*OUT*IN-1
    const int b  = idx >> 16;
    const int oi = idx & 0xFFFF;
    float a[K_EXP];
#pragma unroll
    for (int k = 0; k < K_EXP; ++k) a[k] = att[b * K_EXP + k];
    float s = 0.f;
#pragma unroll
    for (int k = 0; k < K_EXP; ++k) s += a[k] * weight[k * (OUT_C * IN_C) + oi];
    aggw[idx] = (_Float16)s;
    if ((oi & (IN_C - 1)) == 0) {
        const int o = oi >> 8;
        float t = 0.f;
#pragma unroll
        for (int k = 0; k < K_EXP; ++k) t += a[k] * bias_k[k * OUT_C + o];
        aggb[b * OUT_C + o] = t;
    }
}

// ------- Kernel 3: per-batch GEMM out[b] = aggw[b](256x256) * x[b](256x4096) + aggb -------
// Block: 256 threads = 8 waves; block tile 256(M) x 64(N); K loop in steps of 32.
__global__ __launch_bounds__(256)
void moe_gemm_kernel(const float* __restrict__ x,
                     const _Float16* __restrict__ aggw,
                     const float* __restrict__ aggb,
                     float* __restrict__ out) {
    __shared__ __align__(16) _Float16 Al[OUT_C][40];  // 256 rows x 32 K (+8 pad halves)
    __shared__ __align__(16) _Float16 Bl[64][40];     // 64 N x 32 K (+pad), K contiguous

    const int b    = blockIdx.y;
    const int h0   = blockIdx.x * 64;
    const int tid  = threadIdx.x;
    const int wave = tid >> 5;
    const int lane = tid & 31;
    const int m0   = wave * 32;
    const int lhalf = (lane < 16) ? 0 : 1;
    const int l15   = lane & 15;

    const _Float16* Ab = aggw + (size_t)b * (OUT_C * IN_C);
    const float*    Xb = x    + (size_t)b * (IN_C * HLEN);
    const float*    Bb = aggb + b * OUT_C;
    float*          Ob = out  + (size_t)b * (OUT_C * HLEN);

    // Init accumulators with per-row bias (free bias add).
    v8f c[2][4];
#pragma unroll
    for (int si = 0; si < 2; ++si) {
#pragma unroll
        for (int v = 0; v < 8; ++v) {
            const float bv = Bb[m0 + si * 16 + lhalf * 8 + v];
#pragma unroll
            for (int sj = 0; sj < 4; ++sj) c[si][sj][v] = bv;
        }
    }

    // B staging assignment: thread handles k-pair (2 rows) x 4 columns so the two
    // converted halves of one column are LDS-adjacent -> single packed b32 store.
    const int bg = tid & 15;         // 0..15 : column group (4 cols), n = bg*4..bg*4+3
    const int bk = (tid >> 4) * 2;   // 0,2,..,30 : even K row pair

    for (int ks = 0; ks < 8; ++ks) {
        const int k0 = ks * 32;
        // Global loads (A: 64B/thread f16; B: 2x16B/thread f32).
        const uint4* ap = (const uint4*)(Ab + tid * IN_C + k0);
        const uint4 a0 = ap[0], a1 = ap[1], a2 = ap[2], a3 = ap[3];
        const float4* xp0 = (const float4*)(Xb + (size_t)(k0 + bk) * HLEN + h0 + bg * 4);
        const float4* xp1 = (const float4*)(Xb + (size_t)(k0 + bk + 1) * HLEN + h0 + bg * 4);
        const float4 r0 = xp0[0];
        const float4 r1 = xp1[0];
        if (ks < 7)
            __builtin_prefetch((const void*)(Xb + (size_t)(k0 + 32 + bk) * HLEN + h0 + bg * 4), 0, 1);

        __syncthreads();  // previous iteration's LDS reads complete
        *(uint4*)&Al[tid][0]  = a0;
        *(uint4*)&Al[tid][8]  = a1;
        *(uint4*)&Al[tid][16] = a2;
        *(uint4*)&Al[tid][24] = a3;
        {
            const int n0 = bg * 4;
            *(v2h*)&Bl[n0 + 0][bk] = (v2h){(_Float16)r0.x, (_Float16)r1.x};
            *(v2h*)&Bl[n0 + 1][bk] = (v2h){(_Float16)r0.y, (_Float16)r1.y};
            *(v2h*)&Bl[n0 + 2][bk] = (v2h){(_Float16)r0.z, (_Float16)r1.z};
            *(v2h*)&Bl[n0 + 3][bk] = (v2h){(_Float16)r0.w, (_Float16)r1.w};
        }
        __syncthreads();

        // A fragments (16x32): lanes 0-15 -> K 0-7 & 16-23, lanes 16-31 -> K 8-15 & 24-31.
        v16h af[2];
#pragma unroll
        for (int si = 0; si < 2; ++si) {
            const _Float16* ar = &Al[m0 + si * 16 + l15][0];
            const v8h lo = *(const v8h*)(ar + lhalf * 8);
            const v8h hi = *(const v8h*)(ar + 16 + lhalf * 8);
            af[si] = __builtin_shufflevector(lo, hi, 0,1,2,3,4,5,6,7,8,9,10,11,12,13,14,15);
        }
        // B fragments (32x16): lanes 0-15 -> K 0-7 & 16-23, lanes 16-31 -> K 8-15 & 24-31 (K contiguous in LDS).
        v16h bf[4];
#pragma unroll
        for (int sj = 0; sj < 4; ++sj) {
            const _Float16* br = &Bl[sj * 16 + l15][0];
            const v8h lo = *(const v8h*)(br + lhalf * 8);
            const v8h hi = *(const v8h*)(br + 16 + lhalf * 8);
            bf[sj] = __builtin_shufflevector(lo, hi, 0,1,2,3,4,5,6,7,8,9,10,11,12,13,14,15);
        }
#pragma unroll
        for (int si = 0; si < 2; ++si)
#pragma unroll
            for (int sj = 0; sj < 4; ++sj)
                c[si][sj] = __builtin_amdgcn_wmma_f32_16x16x32_f16(
                    false, af[si], false, bf[sj], (short)0, c[si][sj], false, false);
    }

    // Epilogue: C layout -> VGPR v is row v (lanes 0-15) / v+8 (lanes 16-31), N = lane&15.
#pragma unroll
    for (int si = 0; si < 2; ++si) {
#pragma unroll
        for (int v = 0; v < 8; ++v) {
            const int row = m0 + si * 16 + lhalf * 8 + v;
            float* orow = Ob + (size_t)row * HLEN + h0 + l15;
#pragma unroll
            for (int sj = 0; sj < 4; ++sj) orow[sj * 16] = c[si][sj][v];
        }
    }
}

extern "C" void kernel_launch(void* const* d_in, const int* in_sizes, int n_in,
                              void* d_out, int out_size, void* d_ws, size_t ws_size,
                              hipStream_t stream) {
    const float* x       = (const float*)d_in[0];
    const float* emb     = (const float*)d_in[1];
    const float* W0      = (const float*)d_in[2];
    const float* b0      = (const float*)d_in[3];
    const float* W1      = (const float*)d_in[4];
    const float* b1      = (const float*)d_in[5];
    const float* W2      = (const float*)d_in[6];
    const float* b2      = (const float*)d_in[7];
    const float* weight  = (const float*)d_in[8];
    const float* bias_k  = (const float*)d_in[9];
    float* out = (float*)d_out;

    // Workspace layout: att (1KB) | aggb (32KB) | [64KB-aligned] aggw f16 (4MB)
    float*    att  = (float*)d_ws;
    float*    aggb = (float*)d_ws + 256;
    _Float16* aggw = (_Float16*)((char*)d_ws + 65536);

    routing_kernel<<<dim3(BS), dim3(HID), 0, stream>>>(emb, W0, b0, W1, b1, W2, b2, att);
    agg_kernel<<<dim3((BS * OUT_C * IN_C) / 256), dim3(256), 0, stream>>>(att, weight, bias_k, aggw, aggb);
    moe_gemm_kernel<<<dim3(HLEN / 64, BS), dim3(256), 0, stream>>>(x, aggw, aggb, out);
}
